// DepthNet_154618823337
// MI455X (gfx1250) — matrix operand order repaired and checked by reading, not compiled, not run on hardware
//
#include <hip/hip_runtime.h>
#include <hip/hip_bf16.h>
#include <math.h>

typedef _Float16 v16h __attribute__((ext_vector_type(16)));
typedef float    v8f  __attribute__((ext_vector_type(8)));
typedef _Float16 h8   __attribute__((ext_vector_type(8)));

namespace {
constexpr int V_ = 3;
constexpr int C_ = 32;
constexpr int D_ = 96;
constexpr int H_ = 128;
constexpr int W_ = 160;
constexpr int HW_ = H_ * W_;
constexpr int VIEW_STRIDE = C_ * HW_;   // per-view feature stride (B=1)
constexpr int WGROUPS = W_ / 16;        // 10
}

// ---------------------------------------------------------------------------
// Kernel 0: fuse projections, invert ref (affine), produce rot(9)+trans(3)
// per src view into RT[(v-1)*12 ...]. Single-thread scalar math.
// ---------------------------------------------------------------------------
__global__ void proj_setup_kernel(const float* __restrict__ P, float* __restrict__ RT)
{
    if (threadIdx.x != 0 || blockIdx.x != 0) return;
    float fused[V_][4][4];
    for (int v = 0; v < V_; ++v) {
        const float* E = P + v * 32;        // proj[0][v][0]
        const float* K = P + v * 32 + 16;   // proj[0][v][1]
        for (int i = 0; i < 3; ++i)
            for (int j = 0; j < 4; ++j)
                fused[v][i][j] = K[i*4+0]*E[0*4+j] + K[i*4+1]*E[1*4+j] + K[i*4+2]*E[2*4+j];
        for (int j = 0; j < 4; ++j) fused[v][3][j] = E[12 + j];
    }
    // affine inverse of fused[0] (last row is [0,0,0,1] by construction)
    float a[3][3], b[3];
    for (int i = 0; i < 3; ++i) { for (int j = 0; j < 3; ++j) a[i][j] = fused[0][i][j]; b[i] = fused[0][i][3]; }
    float c00 =  a[1][1]*a[2][2] - a[1][2]*a[2][1];
    float c01 = -(a[1][0]*a[2][2] - a[1][2]*a[2][0]);
    float c02 =  a[1][0]*a[2][1] - a[1][1]*a[2][0];
    float det = a[0][0]*c00 + a[0][1]*c01 + a[0][2]*c02;
    float id  = 1.0f / det;
    float ai[3][3];
    ai[0][0] = c00*id;
    ai[0][1] = -(a[0][1]*a[2][2] - a[0][2]*a[2][1])*id;
    ai[0][2] =  (a[0][1]*a[1][2] - a[0][2]*a[1][1])*id;
    ai[1][0] = c01*id;
    ai[1][1] =  (a[0][0]*a[2][2] - a[0][2]*a[2][0])*id;
    ai[1][2] = -(a[0][0]*a[1][2] - a[0][2]*a[1][0])*id;
    ai[2][0] = c02*id;
    ai[2][1] = -(a[0][0]*a[2][1] - a[0][1]*a[2][0])*id;
    ai[2][2] =  (a[0][0]*a[1][1] - a[0][1]*a[1][0])*id;
    float inv[4][4];
    for (int i = 0; i < 3; ++i) {
        for (int j = 0; j < 3; ++j) inv[i][j] = ai[i][j];
        inv[i][3] = -(ai[i][0]*b[0] + ai[i][1]*b[1] + ai[i][2]*b[2]);
    }
    inv[3][0] = 0.f; inv[3][1] = 0.f; inv[3][2] = 0.f; inv[3][3] = 1.f;
    for (int v = 1; v < V_; ++v) {
        float Pv[4][4];
        for (int i = 0; i < 4; ++i)
            for (int j = 0; j < 4; ++j) {
                float s = 0.f;
                for (int k = 0; k < 4; ++k) s += fused[v][i][k] * inv[k][j];
                Pv[i][j] = s;
            }
        float* o = RT + (v - 1) * 12;
        o[0]=Pv[0][0]; o[1]=Pv[0][1]; o[2]=Pv[0][2];
        o[3]=Pv[1][0]; o[4]=Pv[1][1]; o[5]=Pv[1][2];
        o[6]=Pv[2][0]; o[7]=Pv[2][1]; o[8]=Pv[2][2];
        o[9]=Pv[0][3]; o[10]=Pv[1][3]; o[11]=Pv[2][3];
    }
}

// ---------------------------------------------------------------------------
// Kernel 1: homography warp + variance, output f16 channels-last volume
// var[((d*H+h)*W+w)*32 + c]. One thread per (d,h,w) voxel.
// Outer loop over 4 groups of 8 channels stays rolled (moderate VGPRs ->
// high occupancy for gather-latency hiding); inner 8-channel loop is fully
// unrolled so all vector-element indices are constants (no scratch alloca),
// and each 16B result chunk is stored immediately.
// ---------------------------------------------------------------------------
__global__ __launch_bounds__(256) void build_variance_kernel(
    const float* __restrict__ feat,   // (V,1,C,H,W)
    const float* __restrict__ RT,     // 2 × (rot9, trans3)
    const float* __restrict__ depthv, // (1,D)
    _Float16* __restrict__ var)
{
    int tid = blockIdx.x * 256 + threadIdx.x;
    if (tid >= D_ * HW_) return;
    int w = tid % W_;
    int h = (tid / W_) % H_;
    int d = tid / HW_;
    float depth = depthv[d];
    float fx = (float)w, fy = (float)h;

    int   idxs[2][4];
    float wgts[2][4];
#pragma unroll
    for (int s = 0; s < 2; ++s) {
        const float* rt = RT + s * 12;
        float X = rt[0]*fx + rt[1]*fy + rt[2];
        float Y = rt[3]*fx + rt[4]*fy + rt[5];
        float Z = rt[6]*fx + rt[7]*fy + rt[8];
        float zx = X*depth + rt[9];
        float zy = Y*depth + rt[10];
        float zz = Z*depth + rt[11];
        float rz = 1.0f / zz;
        float px = zx * rz, py = zy * rz;
        float x0 = floorf(px), y0 = floorf(py);
        float wx1 = px - x0, wx0 = 1.0f - wx1;
        float wy1 = py - y0, wy0 = 1.0f - wy1;
        int xi = (int)x0, yi = (int)y0;
#pragma unroll
        for (int k = 0; k < 4; ++k) {
            int dx = k & 1, dy = k >> 1;
            int xx = xi + dx, yy = yi + dy;
            bool ok = (xx >= 0) && (xx < W_) && (yy >= 0) && (yy < H_);
            int xc = min(max(xx, 0), W_ - 1);
            int yc = min(max(yy, 0), H_ - 1);
            idxs[s][k] = yc * W_ + xc;
            wgts[s][k] = (ok ? 1.0f : 0.0f) * (dx ? wx1 : wx0) * (dy ? wy1 : wy0);
        }
    }

    const float inv3 = 1.0f / 3.0f;
    int base_pix = h * W_ + w;
    h8* dst = (h8*)(var + (size_t)tid * C_);
#pragma unroll 1
    for (int g = 0; g < 4; ++g) {
        h8 outv;
#pragma unroll
        for (int j = 0; j < 8; ++j) {
            int c = g * 8 + j;
            float r = feat[c * HW_ + base_pix];        // ref view (v=0)
            float sum = r, sq = r * r;
#pragma unroll
            for (int s = 0; s < 2; ++s) {
                const float* fv = feat + (s + 1) * VIEW_STRIDE + c * HW_;
                float val = fv[idxs[s][0]] * wgts[s][0]
                          + fv[idxs[s][1]] * wgts[s][1]
                          + fv[idxs[s][2]] * wgts[s][2]
                          + fv[idxs[s][3]] * wgts[s][3];
                sum += val;
                sq  += val * val;
            }
            float mean = sum * inv3;
            float v = sq * inv3 - mean * mean;
            outv[j] = (_Float16)v;                     // constant index
        }
        dst[g] = outv;
    }
}

// ---------------------------------------------------------------------------
// Kernel 2: 3x3x3 x 32ch -> 1ch conv as 27 WMMA f16 16x16x32 ops per wave.
//   A (16x32): row M = variance for pixel wbase+M, K = 32 channels.
//       Lane layout (16-bit A): lanes 0-15 hold M=lane, K0-7 & K16-23;
//       lanes 16-31 hold M=lane-16, K8-15 & K24-31  -> two 16B loads/lane.
//   B (32x16): weight vector for this tap replicated into all 16 columns.
//       Lane n<16: wl[tap][0..7] & [16..23]; lane n+16: wl[tap][8..15] & [24..31]
//       (uniform per half-wave -> no divergence).
//   D: all 16 columns identical; per 16x16 f32 C/D layout lane 0 holds
//       pixels M=0..7 in acc[0..7], lane 16 holds M=8..15.
// ---------------------------------------------------------------------------
__global__ __launch_bounds__(256) void conv_wmma_kernel(
    const _Float16* __restrict__ var,   // [(d*H+h)*W+w][c] f16
    const float* __restrict__ wgt,      // (1,C,3,3,3)
    const float* __restrict__ bias,     // (1,)
    float* __restrict__ cost)           // [d][h][w] f32
{
    __shared__ alignas(16) _Float16 wl[27 * 32];    // [tap][c]
    int t = threadIdx.x;
    for (int i = t; i < 27 * 32; i += 256) {
        int tap = i >> 5, c = i & 31;
        wl[i] = (_Float16)wgt[c * 27 + tap];
    }
    __syncthreads();

    int lane   = t & 31;
    int waveId = t >> 5;
    int group  = blockIdx.x * 8 + waveId;
    int wg  = group % WGROUPS;
    int tmp = group / WGROUPS;
    int h = tmp % H_;
    int d = tmp / H_;
    int wbase = wg * 16;

    int m  = lane & 15;            // A row -> pixel wbase+m
    int hi = lane >> 4;            // half-wave select
    int kb = hi * 16;              // 0 or 16; unused (B K chunks start at hi*8)
    float bias0 = bias[0];

    v8f acc = {};
#pragma unroll
    for (int kd = 0; kd < 3; ++kd) {
        int dd = d + kd - 1;
#pragma unroll
        for (int kh = 0; kh < 3; ++kh) {
            int hh = h + kh - 1;
            bool rowOK = (dd >= 0) && (dd < D_) && (hh >= 0) && (hh < H_);
            size_t rowbase = ((size_t)dd * H_ + hh) * W_;
#pragma unroll
            for (int kw = 0; kw < 3; ++kw) {
                int tap = (kd * 3 + kh) * 3 + kw;
                // --- A fragment: variance of pixel (dd,hh,ww), channel
                // chunks [cb..cb+7] and [cb+16..cb+23] (16B each), cb=hi*8.
                int ww = wbase + m + kw - 1;
                v16h a = {};
                if (rowOK && ww >= 0 && ww < W_) {
                    const h8* src = (const h8*)(var + (rowbase + ww) * C_);
                    h8 lo = src[hi];        // channels cb .. cb+7
                    h8 up = src[hi + 2];    // channels cb+16 .. cb+23
                    a = __builtin_shufflevector(lo, up,
                        0,1,2,3,4,5,6,7, 8,9,10,11,12,13,14,15);
                }
                // --- B fragment: weights[tap][cb..cb+7] and [cb+16..cb+23], uniform/half-wave.
                const h8* wp = (const h8*)&wl[tap * 32];
                v16h b = __builtin_shufflevector(wp[hi], wp[hi + 2],
                        0,1,2,3,4,5,6,7, 8,9,10,11,12,13,14,15);
                acc = __builtin_amdgcn_wmma_f32_16x16x32_f16(
                    false, a, false, b, (short)0, acc, false, false);
            }
        }
    }
    // Column 0 of D: lane 0 -> pixels 0..7, lane 16 -> pixels 8..15.
    if (m == 0) {
        v8f r;
#pragma unroll
        for (int i = 0; i < 8; ++i) r[i] = acc[i] + bias0;
        v8f* dst = (v8f*)(cost + ((size_t)d * H_ + h) * W_ + wbase + hi * 8);
        *dst = r;
    }
}

// ---------------------------------------------------------------------------
// Kernel 3: softmax over D, depth regression, window-4 confidence.
// One thread per (h,w); strided (coalesced) passes over cost.
// ---------------------------------------------------------------------------
__global__ __launch_bounds__(256) void softmax_depth_kernel(
    const float* __restrict__ cost,
    const float* __restrict__ depthv,
    float* __restrict__ out)
{
    int pix = blockIdx.x * 256 + threadIdx.x;
    if (pix >= HW_) return;

    float m = -3.402823466e38f;
    for (int d = 0; d < D_; ++d)
        m = fmaxf(m, cost[d * HW_ + pix]);

    float sum = 0.f, dnum = 0.f, inum = 0.f;
    for (int d = 0; d < D_; ++d) {
        float e = expf(cost[d * HW_ + pix] - m);
        sum  += e;
        dnum += e * depthv[d];
        inum += e * (float)d;
    }
    float rs    = 1.0f / sum;
    float depth = dnum * rs;
    float idxf  = inum * rs;
    int di = (int)idxf;
    di = min(max(di, 0), D_ - 1);

    float c4 = 0.f;
#pragma unroll
    for (int j = -1; j <= 2; ++j) {
        int dj = di + j;
        if (dj >= 0 && dj < D_)
            c4 += expf(cost[dj * HW_ + pix] - m);
    }
    out[pix]       = depth;
    out[HW_ + pix] = c4 * rs;
}

// ---------------------------------------------------------------------------
extern "C" void kernel_launch(void* const* d_in, const int* in_sizes, int n_in,
                              void* d_out, int out_size, void* d_ws, size_t ws_size,
                              hipStream_t stream)
{
    (void)in_sizes; (void)n_in; (void)out_size; (void)ws_size;
    const float* feat   = (const float*)d_in[0];  // (V,1,C,H,W)
    const float* proj   = (const float*)d_in[1];  // (1,V,2,4,4)
    const float* depthv = (const float*)d_in[2];  // (1,D)
    const float* rw     = (const float*)d_in[3];  // (1,C,3,3,3)
    const float* rb     = (const float*)d_in[4];  // (1,)
    float* out = (float*)d_out;                   // depth(HW) ++ confidence(HW)

    char* ws = (char*)d_ws;
    float*    RT   = (float*)ws;                      // 24 floats
    _Float16* var  = (_Float16*)(ws + 256);           // D*H*W*32 halfs
    size_t varBytes = (size_t)D_ * HW_ * C_ * sizeof(_Float16);
    float*    cost = (float*)(ws + 256 + varBytes);   // D*H*W floats

    proj_setup_kernel<<<1, 32, 0, stream>>>(proj, RT);

    int voxels = D_ * HW_;
    build_variance_kernel<<<(voxels + 255) / 256, 256, 0, stream>>>(feat, RT, depthv, var);

    int groups = D_ * H_ * WGROUPS;   // one wave per 16 output pixels
    conv_wmma_kernel<<<groups / 8, 256, 0, stream>>>(var, rw, rb, cost);

    softmax_depth_kernel<<<(HW_ + 255) / 256, 256, 0, stream>>>(cost, depthv, out);
}
